// Swing_10359461118244
// MI455X (gfx1250) — matrix operand order repaired and checked by reading, not compiled, not run on hardware
//
#include <hip/hip_runtime.h>
#include <hip/hip_bf16.h>

typedef __attribute__((ext_vector_type(16))) __bf16          v16bf;
typedef __attribute__((ext_vector_type(8)))  float           v8f;
typedef __attribute__((ext_vector_type(8)))  unsigned short  u16x8;
typedef __attribute__((ext_vector_type(16))) unsigned short  u16x16;

__device__ __forceinline__ unsigned short f2bf(float f) {
  unsigned int u = __builtin_bit_cast(unsigned int, f);
  u += 0x7FFFu + ((u >> 16) & 1u);            // round-to-nearest-even
  return (unsigned short)(u >> 16);
}
__device__ __forceinline__ float bf2f(unsigned short h) {
  unsigned int u = ((unsigned int)h) << 16;
  return __builtin_bit_cast(float, u);
}

// ---------------------------------------------------------------- weights fp32 -> bf16
__global__ void f32_to_bf16_kernel(const float* __restrict__ in,
                                   unsigned short* __restrict__ out, int n) {
  int i = blockIdx.x * blockDim.x + threadIdx.x;
  if (i < n) out[i] = f2bf(in[i]);
}

// lin_w [96][48] fp32 -> bf16 padded to [96][64] (K 48..63 = 0)
__global__ void pad_linw_kernel(const float* __restrict__ lw,
                                unsigned short* __restrict__ lwbf) {
  int i = blockIdx.x * blockDim.x + threadIdx.x;
  if (i < 96 * 64) {
    int e = i >> 6, cc = i & 63;
    lwbf[i] = (cc < 48) ? f2bf(lw[e * 48 + cc]) : (unsigned short)0;
  }
}

// ---------------------------------------------------------------- patch merge + 48->96 linear, WMMA
// GEMM [65536 x 48(pad 64)] * [48 x 96]; A rows gathered straight into WMMA fragments.
// Writes emb[n][e*4096 + s] (bf16) == the token-major [65536 x 96] view used downstream.
__global__ void patch_embed_wmma_kernel(const float* __restrict__ x,
                                        const unsigned short* __restrict__ lwbf,
                                        const float* __restrict__ lb,
                                        unsigned short* __restrict__ emb) {
  int lane = threadIdx.x & 31;
  int wave = (blockIdx.x * blockDim.x + threadIdx.x) >> 5;
  int row0 = wave << 4;                 // first position of this wave's 16-row tile
  int nsub = lane & 15;
  int hi   = lane >> 4;
  int p    = row0 + nsub;               // this lane's A-row (global position)
  int n    = p >> 12;                   // sample (16 | 4096, so uniform per wave)
  int s    = p & 4095;                  // spatial hp*64+wp

  // faithful raw-reshape gather: j = cc*4096 + s over flat (c,hp,wp,py,px) order
  auto gx = [&](int cc) -> unsigned short {
    int j   = (cc << 12) + s;
    int c   = j >> 16;
    int rem = j & 65535;
    int row = ((rem >> 10) << 2) | ((rem >> 2) & 3);
    int col = (((rem >> 4) & 63) << 2) | (rem & 3);
    return f2bf(x[(size_t)((n * 3 + c) << 16) + (row << 8) + col]);
  };

  // Two 16x32 A fragments (contiguous K = 16*hi + elem, same order as B), K padded 48->64 with zeros.
  v16bf Ak[2];
  {
    u16x16 a0, a1;
    int base0 = hi * 16;                // k-step 0: elems 0..15 = K base0..base0+15
#pragma unroll
    for (int i = 0; i < 8; ++i) { a0[i] = gx(base0 + i); a0[i + 8] = gx(base0 + 8 + i); }
    int base1 = 32 + hi * 16;           // k-step 1: cc 32..47 real (hi=0), 48..63 zero-pad (hi=1)
#pragma unroll
    for (int i = 0; i < 8; ++i) { a1[i] = hi ? (unsigned short)0 : gx(base1 + i); a1[i + 8] = hi ? (unsigned short)0 : gx(base1 + 8 + i); }
    Ak[0] = __builtin_bit_cast(v16bf, a0);
    Ak[1] = __builtin_bit_cast(v16bf, a1);
  }

  // D rows of one lane = 8 consecutive spatial positions -> one packed 16B store per tile
  unsigned short* outb = emb + (size_t)n * 393216 + (row0 & 4095) + hi * 8;
#pragma unroll
  for (int ct = 0; ct < 6; ++ct) {
    int col0 = ct << 4;
    const unsigned short* brow = lwbf + ((col0 + nsub) << 6) + hi * 16;
    u16x16 b0 = *(const u16x16*)(brow);
    u16x16 b1 = *(const u16x16*)(brow + 32);
    v8f acc = {};
    acc = __builtin_amdgcn_wmma_f32_16x16x32_bf16(
        false, Ak[0], false, __builtin_bit_cast(v16bf, b0), (short)0, acc, false, false);
    acc = __builtin_amdgcn_wmma_f32_16x16x32_bf16(
        false, Ak[1], false, __builtin_bit_cast(v16bf, b1), (short)0, acc, false, false);
    float bv = lb[col0 + nsub];
    u16x8 pk;
#pragma unroll
    for (int r = 0; r < 8; ++r) pk[r] = f2bf(acc[r] + bv);
    *(u16x8*)(outb + ((size_t)(col0 + nsub) << 12)) = pk;
  }
}

// ---------------------------------------------------------------- bf16 WMMA GEMM, K=96
// C[r, f] = sum_e A[r, e] * W[f, e] + bias[f];  one wave = 16 rows x all col tiles.
__global__ void gemm96_wmma_kernel(const unsigned short* __restrict__ A,
                                   const unsigned short* __restrict__ W,
                                   const float* __restrict__ bias,
                                   unsigned short* __restrict__ outBf,
                                   float* __restrict__ outF,
                                   int nColTiles, int ldo) {
  int lane = threadIdx.x & 31;
  int wave = (blockIdx.x * blockDim.x + threadIdx.x) >> 5;
  int row0 = wave << 4;
  int nsub = lane & 15;
  int hi   = lane >> 4;

  // A tile (16x32 bf16) per K-step: lane<16 row M=lane holds K[0..16);
  // lane>=16 same row holds K[16..32)  (contiguous, matches B)
  v16bf Ak[3];
  const unsigned short* arow = A + (size_t)(row0 + nsub) * 96 + hi * 16;
#pragma unroll
  for (int k = 0; k < 3; ++k) {
    u16x8 lo  = *(const u16x8*)(arow + k * 32);
    u16x8 hi8 = *(const u16x8*)(arow + k * 32 + 8);
    u16x16 cat = __builtin_shufflevector(lo, hi8, 0, 1, 2, 3, 4, 5, 6, 7,
                                         8, 9, 10, 11, 12, 13, 14, 15);
    Ak[k] = __builtin_bit_cast(v16bf, cat);
  }

  for (int ct = 0; ct < nColTiles; ++ct) {
    int col0 = ct << 4;
    // B tile (32x16 KxN): lane = column f, contiguous 16-K run starting at 16*(lane>>4).
    // B = W^T, so per-lane data = a contiguous slice of W's row f. 32B aligned loads.
    const unsigned short* brow = W + (size_t)(col0 + nsub) * 96 + hi * 16;
    if (ct + 1 < nColTiles)
      __builtin_prefetch(brow + 16 * 96, 0, 1);   // next col tile -> global_prefetch_b8
    v8f acc = {};
#pragma unroll
    for (int k = 0; k < 3; ++k) {
      u16x16 b = *(const u16x16*)(brow + k * 32);
      acc = __builtin_amdgcn_wmma_f32_16x16x32_bf16(
          false, Ak[k], false, __builtin_bit_cast(v16bf, b),
          (short)0, acc, false, false);
    }
    float bv = bias[col0 + nsub];
    if (outF) {
      float* orow = outF + (size_t)(row0 + hi * 8) * ldo + col0 + nsub;
#pragma unroll
      for (int r = 0; r < 8; ++r) orow[(size_t)r * ldo] = acc[r] + bv;
    } else {
      unsigned short* orow = outBf + (size_t)(row0 + hi * 8) * ldo + col0 + nsub;
#pragma unroll
      for (int r = 0; r < 8; ++r) orow[(size_t)r * ldo] = f2bf(acc[r] + bv);
    }
  }
}

// ---------------------------------------------------------------- attention across N=16 samples
// head_dim=1: for each (token m, head h): softmax_t(q_s*k_t) @ v. One block per m.
__global__ void attn_kernel(const unsigned short* __restrict__ qkv,
                            unsigned short* __restrict__ aout) {
  __shared__ unsigned int sqk[2304];   // 16 x 288 bf16
  __shared__ unsigned int sob[768];    // 16 x 96 bf16
  int m = blockIdx.x;
  int tid = threadIdx.x;               // 128 threads
  const unsigned int* g = (const unsigned int*)qkv;
#pragma unroll
  for (int it = 0; it < 18; ++it) {    // 2304 = 18*128
    int i = tid + (it << 7);
    int n = i / 144, u = i - n * 144;
    sqk[i] = g[(size_t)(n * 4096 + m) * 144 + u];
  }
  __syncthreads();
  if (tid < 96) {
    const unsigned short* s = (const unsigned short*)sqk;
    float q[16], k[16], v[16];
#pragma unroll
    for (int n = 0; n < 16; ++n) {
      q[n] = bf2f(s[n * 288 + tid]);
      k[n] = bf2f(s[n * 288 + 96 + tid]);
      v[n] = bf2f(s[n * 288 + 192 + tid]);
    }
    unsigned short* so = (unsigned short*)sob;
#pragma unroll
    for (int si = 0; si < 16; ++si) {
      float qs = q[si];
      float sc[16];
      float mx = -3.4e38f;
#pragma unroll
      for (int t = 0; t < 16; ++t) { sc[t] = qs * k[t]; mx = fmaxf(mx, sc[t]); }
      float sum = 0.f, acc = 0.f;
#pragma unroll
      for (int t = 0; t < 16; ++t) {
        float e = __expf(sc[t] - mx);
        sum += e;
        acc += e * v[t];
      }
      so[si * 96 + tid] = f2bf(acc / sum);
    }
  }
  __syncthreads();
  unsigned int* go = (unsigned int*)aout;
#pragma unroll
  for (int it = 0; it < 6; ++it) {     // 768 = 6*128
    int i = tid + (it << 7);
    int n = i / 48, u = i - n * 48;
    go[(size_t)(n * 4096 + m) * 48 + u] = sob[i];
  }
}

extern "C" void kernel_launch(void* const* d_in, const int* in_sizes, int n_in,
                              void* d_out, int out_size, void* d_ws, size_t ws_size,
                              hipStream_t stream) {
  const float* x  = (const float*)d_in[0];
  const float* lw = (const float*)d_in[1];
  const float* lb = (const float*)d_in[2];
  const float* w1 = (const float*)d_in[3];
  const float* b1 = (const float*)d_in[4];
  const float* w2 = (const float*)d_in[5];
  const float* b2 = (const float*)d_in[6];

  char* ws = (char*)d_ws;
  // emb (12.58 MB) is reused as attention output after GEMM1 consumes it.
  unsigned short* emb  = (unsigned short*)(ws);                 // 16*393216 bf16
  unsigned short* qkv  = (unsigned short*)(ws + 12582912);      // 65536*288 bf16
  unsigned short* w1bf = (unsigned short*)(ws + 50331648);      // 288*96 bf16
  unsigned short* w2bf = (unsigned short*)(ws + 50386944);      // 96*96 bf16
  unsigned short* lwbf = (unsigned short*)(ws + 50405376);      // 96*64 bf16 (padded)

  f32_to_bf16_kernel<<<108, 256, 0, stream>>>(w1, w1bf, 288 * 96);
  f32_to_bf16_kernel<<<36, 256, 0, stream>>>(w2, w2bf, 96 * 96);
  pad_linw_kernel<<<24, 256, 0, stream>>>(lw, lwbf);
  // Patch embed: [65536 x 48(->64)] * [48 x 96] on WMMA, 4096 waves
  patch_embed_wmma_kernel<<<512, 256, 0, stream>>>(x, lwbf, lb, emb);
  // QKV: [65536 x 96] * [96 x 288]^T  -> 4096 waves, 18 col tiles each
  gemm96_wmma_kernel<<<512, 256, 0, stream>>>(emb, w1bf, b1, qkv, nullptr, 18, 288);
  attn_kernel<<<4096, 128, 0, stream>>>(qkv, emb /* attn-out reuses emb */);
  // Out-proj: [65536 x 96] * [96 x 96]^T -> fp32 straight into d_out (row-major match)
  gemm96_wmma_kernel<<<512, 256, 0, stream>>>(emb, w2bf, b2, nullptr, (float*)d_out, 6, 96);
}
